// DispInit_87857851007297
// MI455X (gfx1250) — matrix operand order repaired and checked by reading, not compiled, not run on hardware
//
#include <hip/hip_runtime.h>
#include <stdint.h>

#define BS 4
#define NC 128
#define H  96
#define W  320
#define WP 321
#define NBH (BS * H)        // 384 (b,h) pairs
#define HW (H * W)          // 30720
#define CV_STRIDE (W * W)   // 102400

#define MB  160             // GEMM M-block
#define LDA 136             // LDS A row stride (ushorts), padded, 16B-aligned rows
#define LDB 328             // LDS B row stride (ushorts), 656B = odd*16B

#define LOGC     6.4614682f     // log(2*W) = log(640)
#define LOG_HALF -0.69314718f   // log(0.5) (dustbin marginal)

typedef __attribute__((ext_vector_type(16))) __bf16       v16bf;
typedef __attribute__((ext_vector_type(8)))  float        v8f;
typedef __attribute__((ext_vector_type(4)))  unsigned int v4u;
typedef int v4i_gv __attribute__((vector_size(4 * sizeof(int))));  // matches builtin

union FragAB { v4u u[2]; v16bf bf; };

// --- gfx1250 async global->LDS staging (ASYNCcnt path), with fallback -------
// Builtin prototype (from compiler diagnostic):
//   void __builtin_amdgcn_global_load_async_to_lds_b128(
//       v4i_gv __attribute__((address_space(1)))* src_global,
//       v4i_gv __attribute__((address_space(3)))* dst_lds,
//       imm int offset, imm int cpol);
#if __has_builtin(__builtin_amdgcn_global_load_async_to_lds_b128)
#define ASYNC_COPY16(dst_lds, src_glob)                                        \
  __builtin_amdgcn_global_load_async_to_lds_b128(                              \
      (__attribute__((address_space(1))) v4i_gv*)(src_glob),                   \
      (__attribute__((address_space(3))) v4i_gv*)(dst_lds), 0, 0)
#else
#define ASYNC_COPY16(dst_lds, src_glob)                                        \
  (*(v4u*)(dst_lds) = *(const v4u*)(src_glob))
#endif

__device__ __forceinline__ void async_drain() {
#if __has_builtin(__builtin_amdgcn_global_load_async_to_lds_b128)
#if __has_builtin(__builtin_amdgcn_s_wait_asynccnt)
  __builtin_amdgcn_s_wait_asynccnt(0);
#elif defined(__gfx1250__)
  asm volatile("s_wait_asynccnt 0" ::: "memory");
#endif
#endif
}

__device__ __forceinline__ unsigned short f2bf(float x) {
  unsigned int u = __float_as_uint(x);
  u += 0x7fffu + ((u >> 16) & 1u);        // round-to-nearest-even
  return (unsigned short)(u >> 16);
}

// ---------------------------------------------------------------------------
// Kernel 1: LayerNorm over C (NCHW input), emit bf16:
//   n in [0,4):  f0b  [b][h][w][c]  (row-major K for WMMA A fragments)
//   n in [4,8):  f1t  [b][h][c][w]  (pre-transposed: K rows for WMMA B frags)
// ---------------------------------------------------------------------------
__global__ __launch_bounds__(320)
void ln_kernel(const float* __restrict__ feat, const float* __restrict__ lw,
               const float* __restrict__ lb, unsigned short* __restrict__ f0b,
               unsigned short* __restrict__ f1t) {
  __shared__ float slw[NC], slb[NC];
  int tid = threadIdx.x;
  if (tid < NC) { slw[tid] = lw[tid]; slb[tid] = lb[tid]; }
  __syncthreads();

  int n = blockIdx.x / H, h = blockIdx.x % H, w = tid;
  size_t base = (size_t)n * NC * HW + (size_t)h * W + w;   // feat[n][c][h][w]

  float s1 = 0.f, s2 = 0.f;
  for (int c = 0; c < NC; ++c) {
    float x = feat[base + (size_t)c * HW];                 // coalesced across w
    s1 += x; s2 += x * x;
  }
  float mean = s1 * (1.f / NC);
  float var  = fmaxf(s2 * (1.f / NC) - mean * mean, 0.f);
  float inv  = rsqrtf(var + 1e-5f);

  if (n < BS) {
    size_t ob = (((size_t)n * H + h) * W + w) * NC;        // [b][h][w][c]
    union { unsigned short s[8]; v4u v; } pk;
    for (int c0 = 0; c0 < NC; c0 += 8) {
#pragma unroll
      for (int k = 0; k < 8; ++k) {
        int c = c0 + k;
        float x = feat[base + (size_t)c * HW];
        pk.s[k] = f2bf((x - mean) * inv * slw[c] + slb[c]);
      }
      *(v4u*)(&f0b[ob + c0]) = pk.v;                        // 16B store
    }
  } else {
    int b = n - BS;
    size_t ob = ((size_t)b * H + h) * (size_t)(NC * W) + w; // [b][h][c][w]
    for (int c = 0; c < NC; ++c) {
      float x = feat[base + (size_t)c * HW];
      f1t[ob + (size_t)c * W] = f2bf((x - mean) * inv * slw[c] + slb[c]); // coalesced
    }
  }
}

// ---------------------------------------------------------------------------
// Kernel 2: cv[b,h,i,j] = <f0[i,:], f1[j,:]>  via v_wmma_f32_16x16x32_bf16.
// One WG per (b,h,mhalf): A tile 160x128 + full B 128x320 in LDS (~127 KB,
// 2 WGs/WGP). Staging via GLOBAL_LOAD_ASYNC_TO_LDS_B128 (ASYNCcnt).
// Each wave task = 1 M-tile x 4 N-tiles: A frag reused 4x per K-chunk and
// the 4 WMMAs per K-chunk hit independent accumulators (no D->C chaining
// bubbles inside a K-chunk).
// ---------------------------------------------------------------------------
__global__ __launch_bounds__(256)
void cv_wmma_kernel(const unsigned short* __restrict__ f0b,
                    const unsigned short* __restrict__ f1t,
                    float* __restrict__ cv) {
  __shared__ unsigned short sA[MB * LDA];   // 43,520 B
  __shared__ unsigned short sB[NC * LDB];   // 83,968 B

  int bh = blockIdx.y, mh = blockIdx.x;
  int tid = threadIdx.x;
  const unsigned short* gA = f0b + (size_t)bh * (W * NC) + (size_t)mh * MB * NC;
  const unsigned short* gB = f1t + (size_t)bh * (NC * W);

  // Stage A (160 rows x 128 K) and B (128 K-rows x 320 N) as 16B chunks.
  for (int idx = tid; idx < MB * (NC / 8); idx += 256) {
    int r = idx >> 4, ch = idx & 15;
    ASYNC_COPY16(&sA[r * LDA + ch * 8], &gA[r * NC + ch * 8]);
  }
  for (int idx = tid; idx < NC * (W / 8); idx += 256) {
    int r = idx / 40, ch = idx % 40;
    ASYNC_COPY16(&sB[r * LDB + ch * 8], &gB[r * W + ch * 8]);
  }
  async_drain();
  __syncthreads();

  int lane = tid & 31, wave = tid >> 5;
  int half = lane >> 4, l16 = lane & 15;
  float* cvbh = cv + (size_t)bh * CV_STRIDE;

  // 10 M-tiles x 5 N-groups (4 N-tiles each) = 50 tasks over 8 waves.
  for (int t = wave; t < (MB / 16) * (W / 64); t += 8) {
    int mt = t / (W / 64), ng = t % (W / 64);
    v8f acc[4];
#pragma unroll
    for (int g = 0; g < 4; ++g) acc[g] = (v8f){0.f,0.f,0.f,0.f,0.f,0.f,0.f,0.f};

#pragma unroll
    for (int kc = 0; kc < NC / 32; ++kc) {
      // A frag (16x32): lane holds row m=l16; K-pairs per ISA layout:
      //   half 0 -> K in [0..7] and [16..23]; half 1 -> [8..15] and [24..31]
      FragAB a;
      int arow = mt * 16 + l16;
      int k0 = kc * 32 + half * 8;
      a.u[0] = *(const v4u*)&sA[arow * LDA + k0];
      a.u[1] = *(const v4u*)&sA[arow * LDA + k0 + 16];
      int brow = kc * 32 + half * 8;        // B frag: lane = N-col, K same order as A
#pragma unroll
      for (int g = 0; g < 4; ++g) {
        FragAB b;
        int n0 = (ng * 4 + g) * 16;
#pragma unroll
        for (int s = 0; s < 16; ++s)
          ((unsigned short*)&b)[s] = sB[(brow + (s & 7) + (s >> 3) * 16) * LDB + n0 + l16];
        acc[g] = __builtin_amdgcn_wmma_f32_16x16x32_bf16(
            false, a.bf, false, b.bf, (short)0, acc[g], false, false);
      }
    }
    // D layout: VGPR r -> M = r + 8*half, N = l16.
    int mbase = mh * MB + mt * 16 + half * 8;
#pragma unroll
    for (int g = 0; g < 4; ++g) {
      int nbase = (ng * 4 + g) * 16 + l16;
#pragma unroll
      for (int r = 0; r < 8; ++r)
        cvbh[(size_t)(mbase + r) * W + nbase] = acc[g][r];
    }
  }
}

// ---------------------------------------------------------------------------
// Kernel 3: log-domain Sinkhorn (10 iters) + argmax/window regression, fused.
// attn[i][j] derived on the fly from cv (triu mask -10000, zero dustbin);
// cv stays L2-resident (157 MB < 192 MB). One WG per (b,h); u,v in LDS.
// ---------------------------------------------------------------------------
__global__ __launch_bounds__(256)
void sinkhorn_kernel(const float* __restrict__ cv, float* __restrict__ disp,
                     float* __restrict__ conf, float* __restrict__ occ) {
  __shared__ float su[WP], sv[WP];
  int bh = blockIdx.x, tid = threadIdx.x;
  int lane = tid & 31, wave = tid >> 5;
  const float* cvbh = cv + (size_t)bh * CV_STRIDE;

  for (int j = tid; j < WP; j += 256) su[j] = 0.f;
  __syncthreads();

  int j0 = tid, j1 = tid + 256;
  bool has1 = (j1 < WP);

  for (int t = 0; t < 10; ++t) {
    // v-update: column LSE over i of (attn[i][j] + u[i]); thread owns cols.
    float m0 = -3.0e38f, s0 = 0.f, m1 = -3.0e38f, s1 = 0.f;
    for (int i = 0; i < WP; ++i) {
      float ui = su[i];                                   // LDS broadcast
      float a0 = (i == W || j0 == W) ? 0.f
               : (j0 > i) ? -10000.f : cvbh[i * W + j0];  // coalesced
      float x = a0 + ui;
      if (x > m0) { s0 = s0 * __expf(m0 - x) + 1.f; m0 = x; }
      else s0 += __expf(x - m0);
      if (has1) {
        float a1 = (i == W || j1 == W) ? 0.f
                 : (j1 > i) ? -10000.f : cvbh[i * W + j1];
        float y = a1 + ui;
        if (y > m1) { s1 = s1 * __expf(m1 - y) + 1.f; m1 = y; }
        else s1 += __expf(y - m1);
      }
    }
    sv[j0] = ((j0 < W) ? -LOGC : LOG_HALF) - (m0 + __logf(fmaxf(s0, 1e-30f)));
    if (has1)
      sv[j1] = ((j1 < W) ? -LOGC : LOG_HALF) - (m1 + __logf(fmaxf(s1, 1e-30f)));
    __syncthreads();

    // u-update: row LSE over j of (attn[i][j] + v[j]); one wave per row.
    for (int i = wave; i < WP; i += 8) {
      float m = -3.0e38f, s = 0.f;
      for (int j = lane; j < WP; j += 32) {
        float a = (i == W || j == W) ? 0.f
                : (j > i) ? -10000.f : cvbh[i * W + j];
        float x = a + sv[j];
        if (x > m) { s = s * __expf(m - x) + 1.f; m = x; }
        else s += __expf(x - m);
      }
#pragma unroll
      for (int off = 16; off > 0; off >>= 1) {            // wave32 butterfly LSE
        float mo = __shfl_xor(m, off, 32);
        float so = __shfl_xor(s, off, 32);
        if (mo > m) { s = s * __expf(m - mo) + so; m = mo; }
        else s += so * __expf(mo - m);
      }
      if (lane == 0)
        su[i] = ((i < W) ? -LOGC : LOG_HALF) - (m + __logf(fmaxf(s, 1e-30f)));
    }
    __syncthreads();
  }

  // Final: prob = exp(attn + u + v + log(2W)); masked_prob zero for j > i.
  int b = bh / H, h = bh % H;
  size_t obase = ((size_t)b * H + h) * W;
  for (int i = wave; i < W; i += 8) {
    float ui = su[i];
    float best = -1.f, occs = 0.f; int bj = 0;
    for (int j = lane; j <= i; j += 32) {
      float p = __expf(cvbh[i * W + j] + ui + sv[j] + LOGC);
      occs += p;
      if (p > best) { best = p; bj = j; }
    }
#pragma unroll
    for (int off = 16; off > 0; off >>= 1) {
      float ob2 = __shfl_xor(best, off, 32);
      int   oj  = __shfl_xor(bj, off, 32);
      occs += __shfl_xor(occs, off, 32);
      if (ob2 > best || (ob2 == best && oj < bj)) { best = ob2; bj = oj; }
    }
    // 5-wide window around argmax (zeros outside [0,i]).
    float wk = 0.f; int jj = bj - 2 + lane;
    if (lane < 5 && jj >= 0 && jj <= i)
      wk = __expf(cvbh[i * W + jj] + ui + sv[jj] + LOGC);
    float cf = wk, cr = wk * (float)jj;
#pragma unroll
    for (int off = 16; off > 0; off >>= 1) {
      cf += __shfl_xor(cf, off, 32);
      cr += __shfl_xor(cr, off, 32);
    }
    cr = (cr + 1e-4f) / (cf + 1e-4f);
    if (lane == 0) {
      disp[obase + i] = (float)i - cr;
      conf[obase + i] = cf;
      occ[obase + i]  = occs;
    }
  }
}

// ---------------------------------------------------------------------------
extern "C" void kernel_launch(void* const* d_in, const int* in_sizes, int n_in,
                              void* d_out, int out_size, void* d_ws, size_t ws_size,
                              hipStream_t stream) {
  (void)in_sizes; (void)n_in; (void)out_size; (void)ws_size;
  const float* feature = (const float*)d_in[0];
  const float* lw = (const float*)d_in[1];
  const float* lb = (const float*)d_in[2];

  float* out  = (float*)d_out;
  float* disp = out;                               // [4,1,96,320]
  float* conf = out + (size_t)BS * H * W;          // [4,1,96,320]
  float* occ  = out + (size_t)2 * BS * H * W;      // [4,1,96,320]
  float* cv   = out + (size_t)3 * BS * H * W;      // [4,96,320,320]

  unsigned short* f0b = (unsigned short*)d_ws;             // bf16 [4][96][320][128]
  unsigned short* f1t = f0b + (size_t)BS * H * W * NC;     // bf16 [4][96][128][320]

  ln_kernel<<<dim3(2 * BS * H), 320, 0, stream>>>(feature, lw, lb, f0b, f1t);
  cv_wmma_kernel<<<dim3(2, NBH), 256, 0, stream>>>(f0b, f1t, cv);
  sinkhorn_kernel<<<dim3(NBH), 256, 0, stream>>>(cv, disp, conf, occ);
}
